// MultiheadCrossAttentionWithRoPE_34239479284092
// MI455X (gfx1250) — matrix operand order repaired and checked by reading, not compiled, not run on hardware
//
#include <hip/hip_runtime.h>
#include <hip/hip_bf16.h>

// ---------------------------------------------------------------------------
// MultiheadCrossAttentionWithRoPE for MI455X (gfx1250, wave32, WMMA)
//   B=4, T=M=2048, D=1024, H=16, hd=64
// All matrix math via v_wmma_f32_16x16x32_bf16 (fp32 accumulate).
// Softmax scale (1/8) and log2(e) folded into Q during RoPE (exp2 domain).
// Cross-lane softmax reductions via v_permlane16_b32 (VALU, no LDS traffic).
// ---------------------------------------------------------------------------

typedef __attribute__((ext_vector_type(8)))  float  v8f;
typedef __attribute__((ext_vector_type(16))) __bf16 v16bf;

#define DIM   1024
#define BT    8192      // B*T == B*M rows
#define SEQ   2048
#define HEADS 16
#define HD    64

// 0.125 (1/sqrt(64)) * log2(e): folded into Q so scores are in exp2 domain
#define QSCALE 0.1803368801f

struct BF16x16 { union { v16bf v; unsigned short u[16]; }; };

static __device__ __forceinline__ unsigned short f2bf(float f) {
    union { float f; unsigned u; } x; x.f = f;
    unsigned r = x.u + 0x7FFFu + ((x.u >> 16) & 1u);   // round-to-nearest-even
    return (unsigned short)(r >> 16);
}
static __device__ __forceinline__ float bf2f(unsigned short b) {
    union { unsigned u; float f; } x; x.u = ((unsigned)b) << 16;
    return x.f;
}
static __device__ __forceinline__ v8f wmma_bf16(v16bf a, v16bf b, v8f c) {
    return __builtin_amdgcn_wmma_f32_16x16x32_bf16(false, a, false, b,
                                                   (short)0, c, false, false);
}

// --- xor-butterfly within 16-lane groups.  v_permlane16_b32 if available
// (pure VALU cross-lane), else __shfl_xor (ds_bpermute).
#if __has_builtin(__builtin_amdgcn_permlane16)
static __device__ __forceinline__ float plane16(float v, unsigned s0, unsigned s1) {
    union { float f; unsigned u; } x; x.f = v;
    x.u = __builtin_amdgcn_permlane16(x.u, x.u, s0, s1, false, false);
    return x.f;
}
static __device__ __forceinline__ float xmax16(float v) {
    v = fmaxf(v, plane16(v, 0x67452301u, 0xEFCDAB89u));  // xor 1
    v = fmaxf(v, plane16(v, 0x54761032u, 0xDCFE98BAu));  // xor 2
    v = fmaxf(v, plane16(v, 0x32107654u, 0xBA98FEDCu));  // xor 4
    v = fmaxf(v, plane16(v, 0xFEDCBA98u, 0x76543210u));  // xor 8
    return v;
}
static __device__ __forceinline__ float xsum16(float v) {
    v = v + plane16(v, 0x67452301u, 0xEFCDAB89u);
    v = v + plane16(v, 0x54761032u, 0xDCFE98BAu);
    v = v + plane16(v, 0x32107654u, 0xBA98FEDCu);
    v = v + plane16(v, 0xFEDCBA98u, 0x76543210u);
    return v;
}
#else
static __device__ __forceinline__ float xmax16(float v) {
    #pragma unroll
    for (int msk = 1; msk < 16; msk <<= 1)
        v = fmaxf(v, __shfl_xor(v, msk, 32));
    return v;
}
static __device__ __forceinline__ float xsum16(float v) {
    #pragma unroll
    for (int msk = 1; msk < 16; msk <<= 1)
        v += __shfl_xor(v, msk, 32);
    return v;
}
#endif

// ---------------------------------------------------------------------------
// Tiled GEMM: out[row, col] = sum_k A[row,k] * W[col,k] + bias[col]
// A fp32 [BT x 1024], W fp32 [1024 x 1024] (used transposed), out bf16 or f32.
// Block tile 128x128, 8 waves (4x2), wave tile 32x64:
//   2 A-frags x 4 B-frags = 8 wmma per 32-wide k-step; each B-frag feeds 2 wmma.
// ---------------------------------------------------------------------------
template <bool BF16OUT>
__global__ void __launch_bounds__(256)
proj_gemm(const float* __restrict__ A, const float* __restrict__ W,
          const float* __restrict__ bias, void* __restrict__ outp)
{
    __shared__ unsigned short lA[128 * 40];   // [m][k], pad stride 40
    __shared__ unsigned short lBt[32 * 136];  // [k][n], pad stride 136

    const int tid  = threadIdx.x;
    const int lane = tid & 31;
    const int wave = tid >> 5;
    const int wr   = wave >> 1;          // 0..3: 32-row slice
    const int wc   = wave & 1;           // 0..1: 64-col slice
    const int rowBase = blockIdx.x * 128;
    const int colBase = blockIdx.y * 128;

    v8f zero = {0.f,0.f,0.f,0.f,0.f,0.f,0.f,0.f};
    v8f acc[2][4];
    #pragma unroll
    for (int rg = 0; rg < 2; ++rg)
        #pragma unroll
        for (int g = 0; g < 4; ++g) acc[rg][g] = zero;

    const int m  = lane & 15;
    const int kb = (lane >> 4) * 8;

    for (int kk = 0; kk < DIM; kk += 32) {
        // --- stage A (128x32 fp32 -> bf16 LDS), 1024 float4 chunks / 256 thr
        #pragma unroll
        for (int i = 0; i < 4; ++i) {
            int c = i * 256 + tid;
            int r = c >> 3, c4 = (c & 7) * 4;
            float4 f = *(const float4*)&A[(size_t)(rowBase + r) * DIM + kk + c4];
            lA[r*40 + c4 + 0] = f2bf(f.x);
            lA[r*40 + c4 + 1] = f2bf(f.y);
            lA[r*40 + c4 + 2] = f2bf(f.z);
            lA[r*40 + c4 + 3] = f2bf(f.w);
        }
        // --- stage W transposed (128x32 fp32 -> bf16 LDS [k][n])
        #pragma unroll
        for (int i = 0; i < 4; ++i) {
            int c = i * 256 + tid;
            int n = c >> 3, c4 = (c & 7) * 4;
            float4 f = *(const float4*)&W[(size_t)(colBase + n) * DIM + kk + c4];
            lBt[(c4+0)*136 + n] = f2bf(f.x);
            lBt[(c4+1)*136 + n] = f2bf(f.y);
            lBt[(c4+2)*136 + n] = f2bf(f.z);
            lBt[(c4+3)*136 + n] = f2bf(f.w);
        }
        if (kk + 32 < DIM) {   // global_prefetch_b8 for next k-tile
            __builtin_prefetch(&A[(size_t)(rowBase + (tid >> 1)) * DIM + kk + 32], 0, 3);
            __builtin_prefetch(&W[(size_t)(colBase + (tid >> 1)) * DIM + kk + 32], 0, 3);
        }
        __syncthreads();

        // --- two A fragments: lane = M (mod 16), halves = K pairs
        BF16x16 af[2];
        #pragma unroll
        for (int rg = 0; rg < 2; ++rg) {
            int aRow = wr * 32 + rg * 16 + m;
            #pragma unroll
            for (int j = 0; j < 8; ++j) {
                int k = (j < 4) ? (kb + 2*j) : (16 + kb + 2*(j - 4));
                af[rg].u[2*j]   = lA[aRow*40 + k];
                af[rg].u[2*j+1] = lA[aRow*40 + k + 1];
            }
        }
        // --- 4 N-tiles: B fragment lane = N (mod 16), halves = K pairs; reused 2x
        #pragma unroll
        for (int g = 0; g < 4; ++g) {
            BF16x16 bf;
            int n0 = wc * 64 + g * 16;
            #pragma unroll
            for (int e = 0; e < 16; ++e) bf.u[e] = lBt[((e < 8) ? (kb + e) : (16 + kb + e - 8))*136 + n0 + m];
            acc[0][g] = wmma_bf16(af[0].v, bf.v, acc[0][g]);
            acc[1][g] = wmma_bf16(af[1].v, bf.v, acc[1][g]);
        }
        __syncthreads();
    }

    // --- epilogue: C layout: reg r -> M = r + 8*(lane>>4), N = lane&15
    #pragma unroll
    for (int rg = 0; rg < 2; ++rg) {
        #pragma unroll
        for (int g = 0; g < 4; ++g) {
            int col = colBase + wc*64 + g*16 + m;
            float bb = bias[col];
            #pragma unroll
            for (int r = 0; r < 8; ++r) {
                int row = rowBase + wr*32 + rg*16 + r + ((lane >> 4) << 3);
                float v = acc[rg][g][r] + bb;
                if (BF16OUT)
                    ((unsigned short*)outp)[(size_t)row * DIM + col] = f2bf(v);
                else
                    ((float*)outp)[(size_t)row * DIM + col] = v;
            }
        }
    }
}

// ---------------------------------------------------------------------------
// RoPE on bf16 Q in place, then fold in QSCALE = 0.125*log2(e).
// Q layout [BT, H, 64]; pairs (i, i+32), i<32.
// ---------------------------------------------------------------------------
__global__ void __launch_bounds__(256)
rope_kernel(unsigned short* __restrict__ qbf, const int* __restrict__ pos)
{
    int i = blockIdx.x * blockDim.x + threadIdx.x;   // BT*H*32 pairs
    if (i >= BT * HEADS * 32) return;
    int bt  = i >> 9;          // 16 heads * 32 pairs per token
    int rem = i & 511;
    int h   = rem >> 5;
    int fi  = rem & 31;
    float p   = (float)pos[bt];
    // inv_freq = 10000^(-2*fi/64) = exp(-fi * ln(10000)/32)
    float inv = __expf(-(float)fi * (9.210340372f / 32.0f));
    float ang = p * inv;
    float c = __cosf(ang), s = __sinf(ang);
    size_t base = ((size_t)bt * HEADS + h) * HD;
    float q1 = bf2f(qbf[base + fi]);
    float q2 = bf2f(qbf[base + 32 + fi]);
    qbf[base + fi]      = f2bf((q1 * c - q2 * s) * QSCALE);
    qbf[base + 32 + fi] = f2bf((q2 * c + q1 * s) * QSCALE);
}

// ---------------------------------------------------------------------------
// Flash attention per (b, h): block = 8 waves, each wave owns 16 query rows
// (128 rows/block). 64-key tiles staged in LDS, online softmax in exp2 domain.
// Key order inside a tile is permuted (column 4m+j <-> key 16j+m) identically
// for P and V so each lane's 4 probabilities pack into one ds_store_b64.
// ---------------------------------------------------------------------------
__global__ void __launch_bounds__(256)
attn_kernel(const unsigned short* __restrict__ Q,
            const unsigned short* __restrict__ K,
            const unsigned short* __restrict__ V,
            float* __restrict__ O)
{
    __shared__ unsigned short lKt[64 * 72];       // transposed [hd][key]
    __shared__ unsigned short lV [64 * 72];       // permuted   [key'][hd]
    __shared__ unsigned short lP [8 * 16 * 72];   // per-wave P [row][key']

    const int tid = threadIdx.x, lane = tid & 31, wid = tid >> 5;
    int tchunk = blockIdx.x & 15;                 // T/128 = 16
    int h      = (blockIdx.x >> 4) & 15;
    int b      = blockIdx.x >> 8;
    const int q0 = tchunk * 128 + wid * 16;
    const int m  = lane & 15;
    const int kb = (lane >> 4) * 8;

    // --- Q fragments (k = 0..31 and 32..63); Q is pre-scaled by QSCALE
    BF16x16 qa0, qa1;
    {
        const unsigned short* qp =
            Q + ((size_t)(b * SEQ + q0 + m) * DIM + h * HD);
        #pragma unroll
        for (int j = 0; j < 8; ++j) {
            int k = (j < 4) ? (kb + 2*j) : (16 + kb + 2*(j - 4));
            qa0.u[2*j]   = qp[k];       qa0.u[2*j+1] = qp[k + 1];
            qa1.u[2*j]   = qp[32 + k];  qa1.u[2*j+1] = qp[32 + k + 1];
        }
    }

    v8f zero = {0.f,0.f,0.f,0.f,0.f,0.f,0.f,0.f};
    float mrow[8], lrow[8];
    v8f o[4];
    #pragma unroll
    for (int r = 0; r < 8; ++r) { mrow[r] = -1e30f; lrow[r] = 0.f; }
    #pragma unroll
    for (int g = 0; g < 4; ++g) o[g] = zero;

    for (int mt = 0; mt < SEQ; mt += 64) {
        // --- stage K (transposed) and V (permuted rows): 2 uint4/thread each
        #pragma unroll
        for (int i = 0; i < 2; ++i) {
            int c = i * 256 + tid;
            int keyi = c >> 3, ch = c & 7;
            const unsigned short* kp =
                K + ((size_t)(b * SEQ + mt + keyi) * DIM + h * HD + ch * 8);
            const unsigned short* vp =
                V + ((size_t)(b * SEQ + mt + keyi) * DIM + h * HD + ch * 8);
            uint4 kv = *(const uint4*)kp;
            uint4 vv = *(const uint4*)vp;
            int vrow = ((keyi & 15) << 2) | (keyi >> 4);   // key' permutation
            *(uint4*)&lV[vrow*72 + ch*8] = vv;
            union { uint4 q; unsigned short u[8]; } kc; kc.q = kv;
            #pragma unroll
            for (int j = 0; j < 8; ++j) lKt[(ch*8 + j)*72 + keyi] = kc.u[j];
            if (mt + 64 < SEQ) __builtin_prefetch(kp + 64 * DIM, 0, 3);
        }
        __syncthreads();

        // --- S = Q (16x64) @ K_tile^T (64x64): 4 N-tiles x 2 k-steps
        v8f s[4];
        #pragma unroll
        for (int st = 0; st < 4; ++st) {
            BF16x16 b0, b1;     // B frag: lane = N(key), halves = K(hd) pairs
            #pragma unroll
            for (int e = 0; e < 16; ++e) {
                int k = (e < 8) ? (kb + e) : (16 + kb + e - 8);
                b0.u[e] = lKt[k * 72        + st*16 + m];   // hd = k
                b1.u[e] = lKt[(32 + k) * 72 + st*16 + m];   // hd = 32+k
            }
            v8f t = zero;
            t = wmma_bf16(qa0.v, b0.v, t);
            t = wmma_bf16(qa1.v, b1.v, t);
            s[st] = t;
        }

        // --- online softmax (exp2 domain): rows live in 16-lane half groups
        #pragma unroll
        for (int r = 0; r < 8; ++r) {
            float v0 = s[0][r], v1 = s[1][r], v2 = s[2][r], v3 = s[3][r];
            float tm = xmax16(fmaxf(fmaxf(v0, v1), fmaxf(v2, v3)));
            float mnew = fmaxf(mrow[r], tm);
            float fac  = __builtin_amdgcn_exp2f(mrow[r] - mnew);
            float p0 = __builtin_amdgcn_exp2f(v0 - mnew);
            float p1 = __builtin_amdgcn_exp2f(v1 - mnew);
            float p2 = __builtin_amdgcn_exp2f(v2 - mnew);
            float p3 = __builtin_amdgcn_exp2f(v3 - mnew);
            float rs = xsum16((p0 + p1) + (p2 + p3));
            lrow[r] = lrow[r] * fac + rs;
            mrow[r] = mnew;
            #pragma unroll
            for (int g = 0; g < 4; ++g) o[g][r] *= fac;
            // lane's 4 probs are keys 16j+m -> contiguous at key' = 4m+j
            int prow = r + ((lane >> 4) << 3);
            uint2 pk;
            pk.x = ((unsigned)f2bf(p1) << 16) | (unsigned)f2bf(p0);
            pk.y = ((unsigned)f2bf(p3) << 16) | (unsigned)f2bf(p2);
            *(uint2*)&lP[wid*1152 + prow*72 + 4*m] = pk;   // one b64 store
        }

        // --- reload P as two A fragments (wave-private LDS region)
        BF16x16 pa0, pa1;
        {
            const unsigned short* pp = &lP[wid*1152 + m*72];
            #pragma unroll
            for (int j = 0; j < 8; ++j) {
                int k = (j < 4) ? (kb + 2*j) : (16 + kb + 2*(j - 4));
                pa0.u[2*j]   = pp[k];       pa0.u[2*j+1] = pp[k + 1];
                pa1.u[2*j]   = pp[32 + k];  pa1.u[2*j+1] = pp[32 + k + 1];
            }
        }
        // --- O += P (16x64) @ V_tile (64x64)  (both in permuted key order)
        #pragma unroll
        for (int g = 0; g < 4; ++g) {
            BF16x16 bv0, bv1;  // B frag: lane = N(hd col), halves = K(key') pairs
            #pragma unroll
            for (int e = 0; e < 16; ++e) {
                int k = (e < 8) ? (kb + e) : (16 + kb + e - 8);
                bv0.u[e] = lV[k*72        + g*16 + m];
                bv1.u[e] = lV[(32 + k)*72 + g*16 + m];
            }
            o[g] = wmma_bf16(pa0.v, bv0.v, o[g]);
            o[g] = wmma_bf16(pa1.v, bv1.v, o[g]);
        }
        __syncthreads();
    }

    // --- normalize and write fp32 attention output [BT, D]
    #pragma unroll
    for (int g = 0; g < 4; ++g) {
        #pragma unroll
        for (int r = 0; r < 8; ++r) {
            int trow = q0 + r + ((lane >> 4) << 3);
            float v = o[g][r] / lrow[r];
            O[(size_t)(b * SEQ + trow) * DIM + h * HD + g*16 + m] = v;
        }
    }
}

// ---------------------------------------------------------------------------
extern "C" void kernel_launch(void* const* d_in, const int* in_sizes, int n_in,
                              void* d_out, int out_size, void* d_ws, size_t ws_size,
                              hipStream_t stream)
{
    const float* query = (const float*)d_in[0];
    const float* key   = (const float*)d_in[1];
    const float* value = (const float*)d_in[2];
    const int*   pos   = (const int*)  d_in[3];
    const float* Wq = (const float*)d_in[4];
    const float* bq = (const float*)d_in[5];
    const float* Wk = (const float*)d_in[6];
    const float* bk = (const float*)d_in[7];
    const float* Wv = (const float*)d_in[8];
    const float* bv = (const float*)d_in[9];
    const float* Wo = (const float*)d_in[10];
    const float* bo = (const float*)d_in[11];
    float* out = (float*)d_out;

    // workspace layout: Qbf(16MB) Kbf(16MB) Vbf(16MB) attn_f32(32MB) = 80MB
    char* ws = (char*)d_ws;
    unsigned short* Qbf = (unsigned short*)(ws);
    unsigned short* Kbf = (unsigned short*)(ws + ((size_t)16 << 20));
    unsigned short* Vbf = (unsigned short*)(ws + ((size_t)32 << 20));
    float*          att = (float*)        (ws + ((size_t)48 << 20));

    dim3 gridG(BT / 128, DIM / 128);   // (64, 8)
    proj_gemm<true ><<<gridG, 256, 0, stream>>>(query, Wq, bq, (void*)Qbf);
    proj_gemm<true ><<<gridG, 256, 0, stream>>>(key,   Wk, bk, (void*)Kbf);
    proj_gemm<true ><<<gridG, 256, 0, stream>>>(value, Wv, bv, (void*)Vbf);

    rope_kernel<<<(BT * HEADS * 32) / 256, 256, 0, stream>>>(Qbf, pos);

    attn_kernel<<<4 * HEADS * (SEQ / 128), 256, 0, stream>>>(Qbf, Kbf, Vbf, att);

    proj_gemm<false><<<gridG, 256, 0, stream>>>(att, Wo, bo, (void*)out);
}
